// VLLMPagedFlashAttention_57732950393102
// MI455X (gfx1250) — compile-verified, matrix operand order checked by reading, not run on hardware
//
#include <hip/hip_runtime.h>

typedef __attribute__((ext_vector_type(16))) _Float16 v16h;
typedef __attribute__((ext_vector_type(8)))  _Float16 v8h;
typedef __attribute__((ext_vector_type(8)))  float    v8f;

#define SEQ    2048
#define HD     128
#define NH     32
#define NKV    8
#define GROUP  4
#define BM     32          // queries per wave (2 x 16-row WMMA tiles)
#define BN     32          // keys per iteration

// Load 16 f32 from one row of a [*, HD] row-major matrix into WMMA A/B lane
// layout for a K=32 slice: elements i<8 -> hd0+i, i>=8 -> hd0+16+(i-8).
// Values are scaled by `mul` before f16 conversion (mul=1.0f folds away).
__device__ __forceinline__ v16h load_tile16(const float* __restrict__ rowp, int hd0, float mul) {
  const float4* p0 = reinterpret_cast<const float4*>(rowp + hd0);
  const float4* p1 = reinterpret_cast<const float4*>(rowp + hd0 + 16);
  float4 f0 = p0[0], f1 = p0[1], f2 = p1[0], f3 = p1[1];
  v16h a;
  a[0]  = (_Float16)(f0.x * mul); a[1]  = (_Float16)(f0.y * mul);
  a[2]  = (_Float16)(f0.z * mul); a[3]  = (_Float16)(f0.w * mul);
  a[4]  = (_Float16)(f1.x * mul); a[5]  = (_Float16)(f1.y * mul);
  a[6]  = (_Float16)(f1.z * mul); a[7]  = (_Float16)(f1.w * mul);
  a[8]  = (_Float16)(f2.x * mul); a[9]  = (_Float16)(f2.y * mul);
  a[10] = (_Float16)(f2.z * mul); a[11] = (_Float16)(f2.w * mul);
  a[12] = (_Float16)(f3.x * mul); a[13] = (_Float16)(f3.y * mul);
  a[14] = (_Float16)(f3.z * mul); a[15] = (_Float16)(f3.w * mul);
  return a;
}

__global__ __launch_bounds__(32)
void VLLMPagedFlashAttention_57732950393102_kernel(const float* __restrict__ q,
                                                   const float* __restrict__ k,
                                                   const float* __restrict__ v,
                                                   float* __restrict__ out) {
  __shared__ alignas(16) _Float16 plds[BM][BN];   // 2 KB P staging

  const int lane = threadIdx.x & 31;
  const int l16  = lane & 15;
  const int hl   = lane >> 4;          // 0: lanes 0-15, 1: lanes 16-31
  const int h    = blockIdx.y;         // q head
  const int kvh  = h / GROUP;          // kv head
  const int qb   = blockIdx.x * BM;    // query tile base

  const float scale = 0.08838834764831845f;  // 1/sqrt(128), folded into Q

  // ---- Load Q tile (32 x 128): 2 row-groups x 4 K=32 slices, pre-scaled ----
  v16h qa[2][4];
#pragma unroll
  for (int g = 0; g < 2; ++g) {
    const float* qrow = q + ((size_t)h * SEQ + (size_t)(qb + 16 * g + l16)) * HD;
#pragma unroll
    for (int s = 0; s < 4; ++s)
      qa[g][s] = load_tile16(qrow, 32 * s + hl * 8, scale);
  }

  // ---- Accumulators: O (32 x 128) as 2x8 C-tiles + running max/sum ----
  v8f o[2][8];
#pragma unroll
  for (int g = 0; g < 2; ++g)
#pragma unroll
    for (int t = 0; t < 8; ++t)
      o[g][t] = (v8f){0.f, 0.f, 0.f, 0.f, 0.f, 0.f, 0.f, 0.f};
  float m_i[2][8], l_i[2][8];
#pragma unroll
  for (int g = 0; g < 2; ++g)
#pragma unroll
    for (int j = 0; j < 8; ++j) { m_i[g][j] = -1e30f; l_i[g][j] = 0.f; }

  const float* kbase = k + (size_t)kvh * SEQ * HD;
  const float* vbase = v + (size_t)kvh * SEQ * HD;

  for (int kb = 0; kb < qb + BM; kb += BN) {
    // Prefetch next K block rows (speculative, global_prefetch_b8)
    if (kb + BN < qb + BM)
      __builtin_prefetch(kbase + (size_t)(kb + BN + l16) * HD, 0, 0);

    // ---- S = Q K^T: 2 key subtiles x 2 q-groups of 16x16 f32 scores.
    //      Each K B-operand is loaded/converted once and feeds 2 WMMAs. ----
    float sc[2][2][8];                       // [qgroup][sub][j]
#pragma unroll
    for (int sub = 0; sub < 2; ++sub) {
      const float* krow = kbase + (size_t)(kb + 16 * sub + l16) * HD;
      v8f sacc[2];
      sacc[0] = (v8f){0.f, 0.f, 0.f, 0.f, 0.f, 0.f, 0.f, 0.f};
      sacc[1] = sacc[0];
#pragma unroll
      for (int s = 0; s < 4; ++s) {
        v16h kbm = load_tile16(krow, 32 * s + hl * 8, 1.0f);
#pragma unroll
        for (int g = 0; g < 2; ++g)          // independent accumulators fill hazard slots
          sacc[g] = __builtin_amdgcn_wmma_f32_16x16x32_f16(
              false, qa[g][s], false, kbm, (short)0, sacc[g], false, false);
      }
      const int col = kb + 16 * sub + l16;
#pragma unroll
      for (int g = 0; g < 2; ++g)
#pragma unroll
        for (int j = 0; j < 8; ++j) {
          const int row = qb + 16 * g + j + 8 * hl;
          sc[g][sub][j] = (col <= row) ? sacc[g][j] : -1e30f;
        }
    }

    // ---- Online softmax: per-row reduce across the 16-lane half ----
    float p0[2][8], p1[2][8], alpha[2][8];
#pragma unroll
    for (int g = 0; g < 2; ++g)
#pragma unroll
      for (int j = 0; j < 8; ++j) {
        float rm = fmaxf(sc[g][0][j], sc[g][1][j]);
        rm = fmaxf(rm, __shfl_xor(rm, 1, 32));
        rm = fmaxf(rm, __shfl_xor(rm, 2, 32));
        rm = fmaxf(rm, __shfl_xor(rm, 4, 32));
        rm = fmaxf(rm, __shfl_xor(rm, 8, 32));
        const float mn = fmaxf(m_i[g][j], rm);
        alpha[g][j] = __expf(m_i[g][j] - mn);
        m_i[g][j] = mn;
        p0[g][j] = __expf(sc[g][0][j] - mn);
        p1[g][j] = __expf(sc[g][1][j] - mn);
        float rs = p0[g][j] + p1[g][j];
        rs += __shfl_xor(rs, 1, 32);
        rs += __shfl_xor(rs, 2, 32);
        rs += __shfl_xor(rs, 4, 32);
        rs += __shfl_xor(rs, 8, 32);
        l_i[g][j] = l_i[g][j] * alpha[g][j] + rs;
      }

    // ---- Rescale O by alpha ----
#pragma unroll
    for (int g = 0; g < 2; ++g)
#pragma unroll
      for (int t = 0; t < 8; ++t)
#pragma unroll
        for (int j = 0; j < 8; ++j)
          o[g][t][j] *= alpha[g][j];

    // ---- C-layout P -> LDS (row major 32x32 f16) ----
#pragma unroll
    for (int g = 0; g < 2; ++g)
#pragma unroll
      for (int j = 0; j < 8; ++j) {
        plds[16 * g + j + 8 * hl][l16]      = (_Float16)p0[g][j];
        plds[16 * g + j + 8 * hl][l16 + 16] = (_Float16)p1[g][j];
      }
    asm volatile("s_wait_dscnt 0x0" ::: "memory");

    // ---- Reload P in A-layout: two b128 LDS reads per lane per q-group ----
    v16h pa[2];
#pragma unroll
    for (int g = 0; g < 2; ++g) {
      const v8h* rp = reinterpret_cast<const v8h*>(&plds[16 * g + l16][hl * 8]);
      v8h lo = rp[0];   // cols hl*8 .. hl*8+7
      v8h hi = rp[2];   // cols hl*8+16 .. hl*8+23
#pragma unroll
      for (int i = 0; i < 8; ++i) { pa[g][i] = lo[i]; pa[g][i + 8] = hi[i]; }
    }

    // ---- O += P V: each V B-operand loaded once, feeds 2 WMMAs ----
#pragma unroll
    for (int t = 0; t < 8; ++t) {
      const float* vcol = vbase + (16 * t + l16);
      v16h vb;
#pragma unroll
      for (int i = 0; i < 16; ++i) {
        const int key = kb + (i & 7) + ((i >> 3) << 4) + hl * 8;
        vb[i] = (_Float16)vcol[(size_t)key * HD];
      }
#pragma unroll
      for (int g = 0; g < 2; ++g)
        o[g][t] = __builtin_amdgcn_wmma_f32_16x16x32_f16(
            false, pa[g], false, vb, (short)0, o[g][t], false, false);
    }
  }

  // ---- Normalize and store (coalesced f32 stores) ----
#pragma unroll
  for (int g = 0; g < 2; ++g)
#pragma unroll
    for (int j = 0; j < 8; ++j) {
      const float inv = 1.0f / l_i[g][j];
      const int row = qb + 16 * g + j + 8 * hl;
      float* op = out + ((size_t)h * SEQ + (size_t)row) * HD + l16;
#pragma unroll
      for (int t = 0; t < 8; ++t)
        op[16 * t] = o[g][t][j] * inv;
    }
}

extern "C" void kernel_launch(void* const* d_in, const int* in_sizes, int n_in,
                              void* d_out, int out_size, void* d_ws, size_t ws_size,
                              hipStream_t stream) {
  (void)in_sizes; (void)n_in; (void)out_size; (void)d_ws; (void)ws_size;
  const float* q = (const float*)d_in[0];
  const float* k = (const float*)d_in[1];
  const float* v = (const float*)d_in[2];
  float* out = (float*)d_out;
  dim3 grid(SEQ / BM, NH, 1);
  VLLMPagedFlashAttention_57732950393102_kernel<<<grid, dim3(32, 1, 1), 0, stream>>>(q, k, v, out);
}
